// SwinTransformerBlock_36069135352030
// MI455X (gfx1250) — matrix operand order repaired and checked by reading, not compiled, not run on hardware
//
#include <hip/hip_runtime.h>
#include <hip/hip_bf16.h>

// ---------------------------------------------------------------------------
// Types / helpers for CDNA5 WMMA (wave32, v_wmma_f32_16x16x32_f16)
// ---------------------------------------------------------------------------
typedef __attribute__((ext_vector_type(16))) _Float16 v16h;
typedef __attribute__((ext_vector_type(8)))  _Float16 v8h;
typedef __attribute__((ext_vector_type(8)))  float    v8f;
typedef __attribute__((ext_vector_type(4)))  int      v4i;

// async global->LDS staging (gfx1250), compile-safe detection
#if defined(__has_builtin)
#if __has_builtin(__builtin_amdgcn_global_load_async_to_lds_b128) && \
    __has_builtin(__builtin_amdgcn_s_wait_asynccnt)
#define ATH_ASYNC_LDS 1
#endif
#endif
#ifndef ATH_ASYNC_LDS
#define ATH_ASYNC_LDS 0
#endif
// AS-qualified v4i for the async builtin (param1 is AS1 v4i*, per diagnostics)
typedef __attribute__((address_space(1))) v4i v4i_g;
typedef __attribute__((address_space(3))) v4i v4i_s;

__device__ __forceinline__ v8f f8zero() {
  v8f r;
#pragma unroll
  for (int i = 0; i < 8; ++i) r[i] = 0.f;
  return r;
}
__device__ __forceinline__ v16h h16zero() {
  v16h r;
#pragma unroll
  for (int i = 0; i < 16; ++i) r[i] = (_Float16)0.f;
  return r;
}
// A-fragment: two contiguous 8-halfword chunks (callers use contiguous k = 16h + s, as B does)
__device__ __forceinline__ v16h frag2(const _Float16* p0, const _Float16* p1) {
  v8h a = *(const v8h*)p0;
  v8h b = *(const v8h*)p1;
  v16h r;
#pragma unroll
  for (int i = 0; i < 8; ++i) { r[i] = a[i]; r[i + 8] = b[i]; }
  return r;
}
__device__ __forceinline__ v8f wmma16(v16h a, v16h b, v8f c) {
  return __builtin_amdgcn_wmma_f32_16x16x32_f16(false, a, false, b, (short)0, c,
                                                false, false);
}
__device__ __forceinline__ float wave_sum(float v) {
#pragma unroll
  for (int off = 16; off; off >>= 1) v += __shfl_xor(v, off, 32);
  return v;
}
__device__ __forceinline__ float gelu_exact(float x) {
  return 0.5f * x * (1.f + erff(x * 0.70710678118654752f));
}

// ---------------------------------------------------------------------------
// Problem constants
// ---------------------------------------------------------------------------
#define BB    8
#define CC    256
#define HEADS 8
#define HD    32
#define N49   49
#define NWIN  256
#define NR    100
#define RP    112        // padded ra column stride (WMMA stores land in pad)
#define TOK   100352     // B * H * W = BWIN * 49
#define BWIN  2048       // B * NWIN
#define IMGH  12544      // NWIN * 49
#define SCALE_F 0.17677669529663687f   // 32^-0.5

// ---------------------------------------------------------------------------
// LayerNorm (+optional shift & window partition) -> f16 activations
// ---------------------------------------------------------------------------
template <int SHIFTED>
__global__ void __launch_bounds__(256)
ln_kernel(const float* __restrict__ xin, const float* __restrict__ g,
          const float* __restrict__ bta, _Float16* __restrict__ outH) {
  int w = threadIdx.x >> 5, l = threadIdx.x & 31;
  int t = blockIdx.x * 8 + w;
  size_t src;
  if (SHIFTED) {
    int win = t / 49, p = t % 49;
    int b = win >> 8, wl = win & 255, wh = wl >> 4, ww = wl & 15;
    int i = p / 7, j = p % 7;
    int hs = (wh * 7 + i + 3) % 112, wsr = (ww * 7 + j + 3) % 112;
    src = (size_t)(b * IMGH + hs * 112 + wsr) * CC;
  } else {
    src = (size_t)t * CC;
  }
  float vals[8], s = 0.f, s2 = 0.f;
  const float* px = xin + src + l * 8;
#pragma unroll
  for (int e = 0; e < 8; ++e) {
    float v = px[e]; vals[e] = v; s += v; s2 += v * v;
  }
  s = wave_sum(s); s2 = wave_sum(s2);
  float mean = s * (1.f / 256.f);
  float var  = s2 * (1.f / 256.f) - mean * mean;
  float inv  = rsqrtf(var + 1e-5f);
  _Float16* po = outH + (size_t)t * CC + l * 8;
#pragma unroll
  for (int e = 0; e < 8; ++e) {
    int c = l * 8 + e;
    po[e] = (_Float16)((vals[e] - mean) * inv * g[c] + bta[c]);
  }
}

__global__ void cvt_f16_kernel(const float* __restrict__ in,
                               _Float16* __restrict__ out, int n) {
  int i = blockIdx.x * blockDim.x + threadIdx.x;
  if (i < n) out[i] = (_Float16)in[i];
}

__global__ void zero_f32_kernel(float* p, int n) {
  int i = blockIdx.x * blockDim.x + threadIdx.x;
  if (i < n) p[i] = 0.f;
}

// ---------------------------------------------------------------------------
// Generic tiled f16 WMMA GEMM: out[M,N] = A[M,K] * W[N,K]^T + bias (+GELU/+res)
// 256 threads / 8 waves per 128x128 tile; double-buffered async LDS staging:
// stage k+1's global->LDS copies are in flight while stage k's WMMAs execute.
// ---------------------------------------------------------------------------
template <int STORE_F16, int DO_GELU, int HAS_RES>
__global__ void __launch_bounds__(256)
gemm_f16_kernel(const _Float16* __restrict__ A, const _Float16* __restrict__ W,
                const float* __restrict__ bias, const float* __restrict__ res,
                float* __restrict__ outF, _Float16* __restrict__ outH,
                int M, int N, int K) {
  __shared__ alignas(32) _Float16 sA[2][128 * 32];
  __shared__ alignas(32) _Float16 sB[2][128 * 32];
  const int tid = threadIdx.x, l = tid & 31, w = tid >> 5;
  const int wm = w >> 1, wn = w & 1;
  const int bM = blockIdx.y * 128, bN = blockIdx.x * 128;
  const int lr = tid >> 1, lh = (tid & 1) * 16;
  const int lm = l & 15;
  const int kb = (l < 16) ? 0 : 8, kb2 = (l < 16) ? 0 : 16;

  v8f acc[2][4];
#pragma unroll
  for (int i = 0; i < 2; ++i)
#pragma unroll
    for (int j = 0; j < 4; ++j) acc[i][j] = f8zero();

  auto stage = [&](int sb, int k0s) {
#if ATH_ASYNC_LDS
    __builtin_amdgcn_global_load_async_to_lds_b128(
        (v4i_g*)(A + (size_t)(bM + lr) * K + k0s + lh),
        (v4i_s*)&sA[sb][lr * 32 + lh], 0, 0);
    __builtin_amdgcn_global_load_async_to_lds_b128(
        (v4i_g*)(W + (size_t)(bN + lr) * K + k0s + lh),
        (v4i_s*)&sB[sb][lr * 32 + lh], 0, 0);
#else
    *(v16h*)&sA[sb][lr * 32 + lh] =
        *(const v16h*)(A + (size_t)(bM + lr) * K + k0s + lh);
    *(v16h*)&sB[sb][lr * 32 + lh] =
        *(const v16h*)(W + (size_t)(bN + lr) * K + k0s + lh);
#endif
  };

  stage(0, 0);
  int cur = 0;
  for (int k0 = 0; k0 < K; k0 += 32) {
    if (k0 + 32 < K) {
      stage(cur ^ 1, k0 + 32);           // prefetch next tile into other buffer
#if ATH_ASYNC_LDS
      __builtin_amdgcn_s_wait_asynccnt(2);  // stage k done; k+1 still in flight
#endif
    } else {
#if ATH_ASYNC_LDS
      __builtin_amdgcn_s_wait_asynccnt(0);
#endif
    }
    __syncthreads();
    v16h af[2], bf[4];
#pragma unroll
    for (int i = 0; i < 2; ++i) {
      const _Float16* p = &sA[cur][(wm * 32 + i * 16 + lm) * 32];
      af[i] = frag2(p + kb2, p + kb2 + 8);
    }
#pragma unroll
    for (int j = 0; j < 4; ++j)
      bf[j] = *(const v16h*)&sB[cur][(wn * 64 + j * 16 + lm) * 32 + kb2];
#pragma unroll
    for (int i = 0; i < 2; ++i)
#pragma unroll
      for (int j = 0; j < 4; ++j) acc[i][j] = wmma16(af[i], bf[j], acc[i][j]);
    __syncthreads();                     // protect cur before it is restaged
    cur ^= 1;
  }

#pragma unroll
  for (int i = 0; i < 2; ++i) {
#pragma unroll
    for (int j = 0; j < 4; ++j) {
      int col = bN + wn * 64 + j * 16 + lm;
      float bcol = bias[col];
#pragma unroll
      for (int r = 0; r < 8; ++r) {
        int row = bM + wm * 32 + i * 16 + ((l < 16) ? r : r + 8);
        float v = acc[i][j][r] + bcol;
        if (DO_GELU) v = gelu_exact(v);
        if (HAS_RES) v += res[(size_t)row * N + col];
        if (STORE_F16) outH[(size_t)row * N + col] = (_Float16)v;
        else           outF[(size_t)row * N + col] = v;
      }
    }
  }
}

// ---------------------------------------------------------------------------
// Reference branch -> ref_q (B,h,NR,HD) f16 and ref_v^T (B,h,HD,128pad) f16
// ---------------------------------------------------------------------------
__global__ void ref_branch_kernel(const float* __restrict__ x_ref,
                                  const float* __restrict__ w,
                                  const float* __restrict__ bias,
                                  const float* __restrict__ mu,
                                  const float* __restrict__ ls,
                                  _Float16* __restrict__ refq,
                                  _Float16* __restrict__ refvT) {
  int idx = blockIdx.x * blockDim.x + threadIdx.x;
  const int total = BB * CC * 228;   // 200 real m + 28 pad writers
  if (idx >= total) return;
  int m = idx % 228;
  int c = (idx / 228) % CC;
  int b = idx / (228 * CC);
  int h = c >> 5, d = c & 31;
  if (m >= 200) {  // zero-pad vT columns 100..127
    refvT[(((size_t)(b * HEADS + h) * HD + d) * 128) + (m - 100)] = (_Float16)0.f;
    return;
  }
  float acc = bias[m];
  const float* xr = x_ref + (size_t)(b * CC + c) * NR;
  const float* wr = w + (size_t)m * NR;
  for (int n = 0; n < NR; ++n) acc += xr[n] * wr[n];
  if (m < NR) {
    float v = mu[c] + __expf(ls[c]) * acc;
    refq[(((size_t)(b * HEADS + h) * NR + m) * HD) + d] = (_Float16)v;
  } else {
    refvT[(((size_t)(b * HEADS + h) * HD + d) * 128) + (m - NR)] = (_Float16)acc;
  }
}

// ---------------------------------------------------------------------------
// ra[b,h,win*49+n, m(pad RP)] = sum_d q[b_,h,n,d] * ref_q[b,h,m,d]
// stores: waves 0..2 unconditional; wave 3 stores only row 48.
// ---------------------------------------------------------------------------
__global__ void __launch_bounds__(128)
ra_gemm_kernel(const _Float16* __restrict__ qkv, const _Float16* __restrict__ refq,
               float* __restrict__ ra) {
  int bid = blockIdx.x;
  int hh = bid & 7, b_ = bid >> 3;
  int b = b_ >> 8, win = b_ & 255;
  int w = threadIdx.x >> 5, l = threadIdx.x & 31, lm = l & 15;
  int kb = (l < 16) ? 0 : 8, kb2 = (l < 16) ? 0 : 16;
  int n = w * 16 + lm;
  v16h af;
  if (n < N49) {
    const _Float16* p = qkv + (size_t)(b_ * 49 + n) * 768 + hh * 32 + kb2;
    af = frag2(p, p + 8);
  } else af = h16zero();
  size_t rabase = ((size_t)(b * HEADS + hh) * IMGH + win * 49) * RP;
#pragma unroll
  for (int t = 0; t < 7; ++t) {
    int m = t * 16 + lm;
    v16h bf = *(const v16h*)(refq + ((size_t)(b * HEADS + hh) * NR + m) * HD + kb2);
    v8f acc = wmma16(af, bf, f8zero());
    if (w < 3) {
      int rbase = w * 16 + ((l < 16) ? 0 : 8);
#pragma unroll
      for (int r = 0; r < 8; ++r)
        ra[rabase + (size_t)(rbase + r) * RP + m] = acc[r];
    } else if (l < 16) {
      ra[rabase + (size_t)48 * RP + m] = acc[0];
    }
  }
}

// ---------------------------------------------------------------------------
// Conv 3x3 (8->8 ch) over (12544,100) + spatial stats (fused); RP strides
// ---------------------------------------------------------------------------
__global__ void __launch_bounds__(256)
conv_stats_kernel(const float* __restrict__ ra, const float* __restrict__ cw,
                  const float* __restrict__ cb, float* __restrict__ u,
                  float* __restrict__ stats) {
  __shared__ float ls[8 * 18 * 102];
  __shared__ float lw[8 * 8 * 9];
  int bid = blockIdx.x;
  int b = bid / 784, yt = bid % 784;
  int y0 = yt * 16;
  int tid = threadIdx.x;
  for (int i = tid; i < 576; i += 256) lw[i] = cw[i];
  for (int i = tid; i < 8 * 18 * 102; i += 256) {
    int ic = i / (18 * 102);
    int rem = i % (18 * 102);
    int yy = rem / 102, xx = rem % 102;
    int gy = y0 + yy - 1, gx = xx - 1;
    float v = 0.f;
    if (gy >= 0 && gy < IMGH && gx >= 0 && gx < NR)
      v = ra[((size_t)(b * HEADS + ic) * IMGH + gy) * RP + gx];
    ls[i] = v;
  }
  __syncthreads();
  int oc = tid >> 5, l = tid & 31;
  float s = 0.f, s2 = 0.f;
  float bconv = cb[oc];
  for (int px = l; px < 1600; px += 32) {
    int y = px / 100, x = px % 100;
    float a = bconv;
#pragma unroll
    for (int ic = 0; ic < 8; ++ic) {
      const float* wr = &lw[(oc * 8 + ic) * 9];
      const float* l0 = &ls[(ic * 18 + y) * 102 + x];
      a += l0[0] * wr[0] + l0[1] * wr[1] + l0[2] * wr[2]
         + l0[102] * wr[3] + l0[103] * wr[4] + l0[104] * wr[5]
         + l0[204] * wr[6] + l0[205] * wr[7] + l0[206] * wr[8];
    }
    u[((size_t)(b * HEADS + oc) * IMGH + y0 + y) * RP + x] = a;
    s += a; s2 += a * a;
  }
  s = wave_sum(s); s2 = wave_sum(s2);
  if (l == 0) {
    atomicAdd(&stats[(b * HEADS + oc) * 2 + 0], s);
    atomicAdd(&stats[(b * HEADS + oc) * 2 + 1], s2);
  }
}

__global__ void conv_apply_kernel(const float* __restrict__ u,
                                  const float* __restrict__ stats,
                                  float* __restrict__ ra) {
  size_t idx = (size_t)blockIdx.x * blockDim.x + threadIdx.x;
  const size_t total = (size_t)BB * HEADS * IMGH * RP;
  if (idx >= total) return;
  int x = (int)(idx % RP);
  if (x >= NR) return;                   // pad columns untouched
  int bc = (int)(idx / (size_t)(IMGH * RP));
  const float inv_n = 1.f / (float)(IMGH * NR);
  float mean = stats[bc * 2 + 0] * inv_n;
  float var  = stats[bc * 2 + 1] * inv_n - mean * mean;
  float v = (u[idx] - mean) * rsqrtf(var + 1e-5f);
  ra[idx] += gelu_exact(v);
}

// ---------------------------------------------------------------------------
// Row softmax over NR=100 -> f16, K-padded to 128 columns (zeros)
// ---------------------------------------------------------------------------
__global__ void __launch_bounds__(256)
ra_softmax_kernel(const float* __restrict__ ra, _Float16* __restrict__ smra) {
  int row = blockIdx.x * 8 + (threadIdx.x >> 5);
  int l = threadIdx.x & 31;
  const float* pr = ra + (size_t)row * RP;
  float v[4];
#pragma unroll
  for (int j = 0; j < 4; ++j) {
    int m = l + 32 * j;
    v[j] = (m < NR) ? pr[m] : -1e30f;
  }
  float mx = fmaxf(fmaxf(v[0], v[1]), fmaxf(v[2], v[3]));
#pragma unroll
  for (int off = 16; off; off >>= 1) mx = fmaxf(mx, __shfl_xor(mx, off, 32));
  float e[4], s = 0.f;
#pragma unroll
  for (int j = 0; j < 4; ++j) {
    int m = l + 32 * j;
    e[j] = (m < NR) ? __expf(v[j] - mx) : 0.f;
    s += e[j];
  }
  s = wave_sum(s);
  float inv = 1.f / s;
  _Float16* po = smra + (size_t)row * 128;
#pragma unroll
  for (int j = 0; j < 4; ++j) po[l + 32 * j] = (_Float16)(e[j] * inv);
}

// ---------------------------------------------------------------------------
// q_new = softmax(ra) @ ref_v * SCALE -> f16 (b_,h, 64pad rows, HD)
// fully unconditional stores (pad rows produce zeros)
// ---------------------------------------------------------------------------
__global__ void __launch_bounds__(128)
qnew_gemm_kernel(const _Float16* __restrict__ smra,
                 const _Float16* __restrict__ refvT,
                 _Float16* __restrict__ qnewP) {
  int bid = blockIdx.x;
  int hh = bid & 7, b_ = bid >> 3;
  int b = b_ >> 8, win = b_ & 255;
  int w = threadIdx.x >> 5, l = threadIdx.x & 31, lm = l & 15;
  int kb = (l < 16) ? 0 : 8, kb2 = (l < 16) ? 0 : 16;
  size_t arow = (size_t)(b * HEADS + hh) * IMGH + win * 49;
  int n = w * 16 + lm;
  v8f acc[2] = {f8zero(), f8zero()};
#pragma unroll
  for (int kk = 0; kk < 4; ++kk) {
    v16h af;
    if (n < N49) {
      const _Float16* p = smra + (arow + n) * 128 + kk * 32 + kb2;
      af = frag2(p, p + 8);
    } else af = h16zero();
#pragma unroll
    for (int t = 0; t < 2; ++t) {
      int d = t * 16 + lm;
      v16h bf = *(const v16h*)(refvT +
          ((size_t)(b * HEADS + hh) * HD + d) * 128 + kk * 32 + kb2);
      acc[t] = wmma16(af, bf, acc[t]);
    }
  }
  int rbase = w * 16 + ((l < 16) ? 0 : 8);
  size_t obase = (size_t)(b_ * HEADS + hh) * 64;
#pragma unroll
  for (int t = 0; t < 2; ++t) {
    int d = t * 16 + lm;
#pragma unroll
    for (int r = 0; r < 8; ++r)
      qnewP[(obase + rbase + r) * HD + d] = (_Float16)(acc[t][r] * SCALE_F);
  }
}

// ---------------------------------------------------------------------------
// v^T per (window, head): (HD=32, 64-padded m) f16
// ---------------------------------------------------------------------------
__global__ void transpose_v_kernel(const _Float16* __restrict__ qkv,
                                   _Float16* __restrict__ vT) {
  size_t idx = (size_t)blockIdx.x * blockDim.x + threadIdx.x;
  const size_t total = (size_t)BWIN * HEADS * HD * 64;
  if (idx >= total) return;
  int m  = (int)(idx & 63);
  int d  = (int)((idx >> 6) & 31);
  int hh = (int)((idx >> 11) & 7);
  int b_ = (int)(idx >> 14);
  _Float16 v = (_Float16)0.f;
  if (m < N49) v = qkv[(size_t)(b_ * 49 + m) * 768 + 512 + hh * 32 + d];
  vT[idx] = v;
}

// ---------------------------------------------------------------------------
// attn = q_new @ k^T (+RPB +mask) -> row softmax -> f16 (64pad x 64pad)
// fully unconditional stores; invalid logits forced to -1e30 (no NaN escape)
// ---------------------------------------------------------------------------
__global__ void __launch_bounds__(128)
attn_kernel(const _Float16* __restrict__ qnewP, const _Float16* __restrict__ qkv,
            const float* __restrict__ rpb, const int* __restrict__ rpi,
            const float* __restrict__ mask, _Float16* __restrict__ attnP) {
  int bid = blockIdx.x;
  int hh = bid & 7, b_ = bid >> 3;
  int win = b_ & 255;
  int w = threadIdx.x >> 5, l = threadIdx.x & 31, lm = l & 15;
  int kb = (l < 16) ? 0 : 8, kb2 = (l < 16) ? 0 : 16;
  int n = w * 16 + lm;
  const _Float16* pa = qnewP + ((size_t)(b_ * HEADS + hh) * 64 + n) * HD + kb2;
  v16h af = frag2(pa, pa + 8);
  v8f acc[4];
#pragma unroll
  for (int t = 0; t < 4; ++t) {
    int m = t * 16 + lm;
    v16h bf = *(const v16h*)(qkv + (size_t)(b_ * 49 + m) * 768 + 256 + hh * 32 + kb2);
    acc[t] = wmma16(af, bf, f8zero());
  }
  size_t obase = (size_t)(b_ * HEADS + hh) * 64;
#pragma unroll
  for (int r = 0; r < 8; ++r) {
    int rn = w * 16 + ((l < 16) ? r : r + 8);
    float lv[4];
#pragma unroll
    for (int t = 0; t < 4; ++t) {
      int m = t * 16 + lm;
      float v;
      if (rn < N49 && m < N49) {
        v = acc[t][r] + rpb[rpi[rn * 49 + m] * HEADS + hh]
                      + mask[((size_t)win * 49 + rn) * 49 + m];
      } else v = -1e30f;   // accumulator never read for invalid cells
      lv[t] = v;
    }
    float mx = fmaxf(fmaxf(lv[0], lv[1]), fmaxf(lv[2], lv[3]));
#pragma unroll
    for (int off = 8; off; off >>= 1) mx = fmaxf(mx, __shfl_xor(mx, off, 16));
    float s = 0.f;
#pragma unroll
    for (int t = 0; t < 4; ++t) { lv[t] = __expf(lv[t] - mx); s += lv[t]; }
#pragma unroll
    for (int off = 8; off; off >>= 1) s += __shfl_xor(s, off, 16);
    float inv = 1.f / s;
#pragma unroll
    for (int t = 0; t < 4; ++t)
      attnP[(obase + rn) * 64 + t * 16 + lm] = (_Float16)(lv[t] * inv);
  }
}

// ---------------------------------------------------------------------------
// out = attnP @ v  -> token-major f16 (waves 0..2 unconditional, wave3 row48)
// ---------------------------------------------------------------------------
__global__ void __launch_bounds__(128)
out_gemm_kernel(const _Float16* __restrict__ attnP,
                const _Float16* __restrict__ vT, _Float16* __restrict__ outB) {
  int bid = blockIdx.x;
  int hh = bid & 7, b_ = bid >> 3;
  int w = threadIdx.x >> 5, l = threadIdx.x & 31, lm = l & 15;
  int kb = (l < 16) ? 0 : 8, kb2 = (l < 16) ? 0 : 16;
  int n = w * 16 + lm;
  v8f acc[2] = {f8zero(), f8zero()};
#pragma unroll
  for (int kk = 0; kk < 2; ++kk) {
    const _Float16* p = attnP + ((size_t)(b_ * HEADS + hh) * 64 + n) * 64 +
                        kk * 32 + kb2;
    v16h af = frag2(p, p + 8);
#pragma unroll
    for (int t = 0; t < 2; ++t) {
      int d = t * 16 + lm;
      v16h bf = *(const v16h*)(vT + ((size_t)(b_ * HEADS + hh) * HD + d) * 64 +
                               kk * 32 + kb2);
      acc[t] = wmma16(af, bf, acc[t]);
    }
  }
  if (w < 3) {
    int rbase = w * 16 + ((l < 16) ? 0 : 8);
#pragma unroll
    for (int t = 0; t < 2; ++t) {
      int d = t * 16 + lm;
#pragma unroll
      for (int r = 0; r < 8; ++r)
        outB[(size_t)(b_ * 49 + rbase + r) * CC + hh * 32 + d] =
            (_Float16)acc[t][r];
    }
  } else if (l < 16) {
#pragma unroll
    for (int t = 0; t < 2; ++t)
      outB[(size_t)(b_ * 49 + 48) * CC + hh * 32 + t * 16 + lm] =
          (_Float16)acc[t][0];
  }
}

// ---------------------------------------------------------------------------
// window-reverse + unshift + residual: xo[orig] = x[orig] + proj[win-order]
// ---------------------------------------------------------------------------
__global__ void unshift_res_kernel(const float* __restrict__ x,
                                   const float* __restrict__ proj,
                                   float* __restrict__ xo) {
  size_t idx = (size_t)blockIdx.x * blockDim.x + threadIdx.x;
  const size_t total = (size_t)TOK * CC;
  if (idx >= total) return;
  int c = (int)(idx & 255);
  int t = (int)(idx >> 8);
  int win = t / 49, p = t % 49;
  int b = win >> 8, wl = win & 255, wh = wl >> 4, ww = wl & 15;
  int i = p / 7, j = p % 7;
  int hs = (wh * 7 + i + 3) % 112, wsr = (ww * 7 + j + 3) % 112;
  size_t orig = (size_t)(b * IMGH + hs * 112 + wsr) * CC + c;
  xo[orig] = x[orig] + proj[idx];
}

// ---------------------------------------------------------------------------
// Host-side launch
// ---------------------------------------------------------------------------
extern "C" void kernel_launch(void* const* d_in, const int* in_sizes, int n_in,
                              void* d_out, int out_size, void* d_ws, size_t ws_size,
                              hipStream_t stream) {
  const float* x       = (const float*)d_in[0];
  const float* mask    = (const float*)d_in[1];
  const float* x_ref   = (const float*)d_in[2];
  const float* n1g     = (const float*)d_in[3];
  const float* n1b     = (const float*)d_in[4];
  const float* qkv_w   = (const float*)d_in[5];
  const float* qkv_b   = (const float*)d_in[6];
  const float* diff_mu = (const float*)d_in[7];
  const float* diff_ls = (const float*)d_in[8];
  const float* rqk_w   = (const float*)d_in[9];
  const float* rqk_b   = (const float*)d_in[10];
  const float* conv_w  = (const float*)d_in[11];
  const float* conv_b  = (const float*)d_in[12];
  const float* rpb     = (const float*)d_in[13];
  const float* proj_w  = (const float*)d_in[14];
  const float* proj_b  = (const float*)d_in[15];
  const float* n2g     = (const float*)d_in[16];
  const float* n2b     = (const float*)d_in[17];
  const float* fc1_w   = (const float*)d_in[18];
  const float* fc1_b   = (const float*)d_in[19];
  const float* fc2_w   = (const float*)d_in[20];
  const float* fc2_b   = (const float*)d_in[21];
  const int*   rpi     = (const int*)d_in[22];
  float* out = (float*)d_out;
  char* wsb = (char*)d_ws;

  // workspace layout (aliasing across pipeline phases)
  const size_t RA_BYTES = (size_t)BB * HEADS * IMGH * RP * 4;  // 359,661,568
  const size_t OFF_R0 = 0;                         // 51,380,224: xw / h2
  const size_t OFF_R1 = OFF_R0 + 51380224;         // 154,140,672: qkv f16 / proj f32
  const size_t OFF_R2 = OFF_R1 + 154140672;        // 819,200: refq f16
  const size_t OFF_R3 = OFF_R2 + 819200;           // 524,288: refvT f16
  const size_t OFF_R4 = OFF_R3 + 524288;           // 67,108,864: vT f16
  const size_t OFF_R5 = OFF_R4 + 67108864;         // RA: ra f32 / attnP+qnewP+outB
  const size_t OFF_R6 = OFF_R5 + RA_BYTES;         // RA: u f32 / smra / h3
  const size_t OFF_R7 = OFF_R6 + RA_BYTES;         // 512: stats
  const size_t OFF_R8 = OFF_R7 + 512;              // 1,572,864: f16 weights

  _Float16* xw    = (_Float16*)(wsb + OFF_R0);
  _Float16* h2    = (_Float16*)(wsb + OFF_R0);
  _Float16* qkvH  = (_Float16*)(wsb + OFF_R1);
  float*    projF = (float*)   (wsb + OFF_R1);
  _Float16* refq  = (_Float16*)(wsb + OFF_R2);
  _Float16* refvT = (_Float16*)(wsb + OFF_R3);
  _Float16* vT    = (_Float16*)(wsb + OFF_R4);
  float*    ra    = (float*)   (wsb + OFF_R5);
  _Float16* attnP = (_Float16*)(wsb + OFF_R5);                  // 134,217,728
  _Float16* qnewP = (_Float16*)(wsb + OFF_R5 + 134217728);      //  67,108,864
  _Float16* outB  = (_Float16*)(wsb + OFF_R5 + 201326592);      //  51,380,224
  float*    u     = (float*)   (wsb + OFF_R6);
  _Float16* smra  = (_Float16*)(wsb + OFF_R6);
  _Float16* h3    = (_Float16*)(wsb + OFF_R6);
  float*    stats = (float*)   (wsb + OFF_R7);
  _Float16* qkvW16  = (_Float16*)(wsb + OFF_R8);
  _Float16* projW16 = (_Float16*)(wsb + OFF_R8 + 393216);
  _Float16* fc1W16  = (_Float16*)(wsb + OFF_R8 + 524288);
  _Float16* fc2W16  = (_Float16*)(wsb + OFF_R8 + 1048576);

  // weight conversions to f16
  cvt_f16_kernel<<<(196608 + 255) / 256, 256, 0, stream>>>(qkv_w, qkvW16, 196608);
  cvt_f16_kernel<<<(65536 + 255) / 256, 256, 0, stream>>>(proj_w, projW16, 65536);
  cvt_f16_kernel<<<(262144 + 255) / 256, 256, 0, stream>>>(fc1_w, fc1W16, 262144);
  cvt_f16_kernel<<<(262144 + 255) / 256, 256, 0, stream>>>(fc2_w, fc2W16, 262144);

  // LN1 + shift + window partition
  ln_kernel<1><<<TOK / 8, 256, 0, stream>>>(x, n1g, n1b, xw);

  // QKV GEMM (M=100352, N=768, K=256), f16 out
  gemm_f16_kernel<1, 0, 0><<<dim3(768 / 128, TOK / 128), 256, 0, stream>>>(
      xw, qkvW16, qkv_b, nullptr, nullptr, qkvH, TOK, 768, 256);

  // reference branch
  ref_branch_kernel<<<(BB * CC * 228 + 255) / 256, 256, 0, stream>>>(
      x_ref, rqk_w, rqk_b, diff_mu, diff_ls, refq, refvT);

  // ra = q @ ref_k^T
  ra_gemm_kernel<<<BWIN * HEADS, 128, 0, stream>>>(qkvH, refq, ra);

  // 3x (conv3x3 + spatial LN + GELU residual)
  for (int it = 0; it < 3; ++it) {
    zero_f32_kernel<<<1, 128, 0, stream>>>(stats, 128);
    conv_stats_kernel<<<BB * 784, 256, 0, stream>>>(ra, conv_w, conv_b, u, stats);
    conv_apply_kernel<<<(int)(((size_t)BB * HEADS * IMGH * RP + 255) / 256), 256,
                        0, stream>>>(u, stats, ra);
  }

  // softmax over NR + f16 K-padded
  ra_softmax_kernel<<<(BB * HEADS * IMGH) / 8, 256, 0, stream>>>(ra, smra);

  // q_new = softmax(ra) @ ref_v * scale  (64-row padded)
  qnew_gemm_kernel<<<BWIN * HEADS, 128, 0, stream>>>(smra, refvT, qnewP);

  // transpose v for the out-GEMM B side
  transpose_v_kernel<<<(int)(((size_t)BWIN * HEADS * HD * 64 + 255) / 256), 256,
                       0, stream>>>(qkvH, vT);

  // attention logits + bias + mask + softmax -> f16 probs (64x64 padded)
  attn_kernel<<<BWIN * HEADS, 128, 0, stream>>>(qnewP, qkvH, rpb, rpi, mask, attnP);

  // out = attn @ v  (token-major f16)
  out_gemm_kernel<<<BWIN * HEADS, 128, 0, stream>>>(attnP, vT, outB);

  // proj GEMM (f32 out, window order)
  gemm_f16_kernel<0, 0, 0><<<dim3(256 / 128, TOK / 128), 256, 0, stream>>>(
      outB, projW16, proj_b, nullptr, projF, nullptr, TOK, 256, 256);

  // window reverse + unshift + residual -> xo (in d_out)
  unshift_res_kernel<<<(int)(((size_t)TOK * CC + 255) / 256), 256, 0, stream>>>(
      x, projF, out);

  // LN2 (token order)
  ln_kernel<0><<<TOK / 8, 256, 0, stream>>>(out, n2g, n2b, h2);

  // fc1 GEMM + GELU (f16 out)
  gemm_f16_kernel<1, 1, 0><<<dim3(1024 / 128, TOK / 128), 256, 0, stream>>>(
      h2, fc1W16, fc1_b, nullptr, nullptr, h3, TOK, 1024, 256);

  // fc2 GEMM + residual(xo) -> d_out (f32)
  gemm_f16_kernel<0, 0, 1><<<dim3(256 / 128, TOK / 128), 256, 0, stream>>>(
      h3, fc2W16, fc2_b, out, out, nullptr, TOK, 256, 1024);
}
